// AttentionBlock2_76699525972556
// MI455X (gfx1250) — matrix operand order repaired and checked by reading, not compiled, not run on hardware
//
#include <hip/hip_runtime.h>
#include <hip/hip_bf16.h>

typedef _Float16 v16h __attribute__((ext_vector_type(16)));
typedef _Float16 h2   __attribute__((ext_vector_type(2)));
typedef float    v8f  __attribute__((ext_vector_type(8)));
typedef int      v4i  __attribute__((ext_vector_type(4)));

#define B_    4
#define C_    64
#define HW_   128
#define HID_  256
#define NELEM (B_ * C_ * HW_ * HW_)   // 4194304

#if __has_builtin(__builtin_amdgcn_global_load_async_to_lds_b128) && \
    __has_builtin(__builtin_amdgcn_s_wait_asynccnt)
#define USE_ASYNC_LDS 1
#else
#define USE_ASYNC_LDS 0
#endif

// ---------------------------------------------------------------------------
// WMMA fragment helpers (CDNA5 16x16x32 f16 layouts, ISA 7.12.2)
// A (16x32, MxK): lane = M (0..15) | K-half (lane>=16 -> +8); VGPR v packs
//   K = (v>>2)*16 + (lane>>4)*8 + (v&3)*2 , +1
// B (32x16, KxN): lane&15 = N; lanes>=16 hold K+8; VGPR v packs the same K as A
// C/D (16x16 f32): VGPR r: M = r + 8*(lane>>4); N = lane&15
// ---------------------------------------------------------------------------
__device__ inline v16h load_a_frag(const _Float16* __restrict__ src, int ld,
                                   int m0, int k0) {
  const int lane = threadIdx.x & 31;
  const int g = lane >> 4;
  const _Float16* row = src + (m0 + (lane & 15)) * ld + k0;
  v16h a;
#pragma unroll
  for (int v = 0; v < 8; ++v) {
    const int k = ((v >> 2) << 4) + (g << 3) + ((v & 3) << 1);
    h2 p = *(const h2*)(row + k);
    a[2 * v] = p.x;
    a[2 * v + 1] = p.y;
  }
  return a;
}

// srcT is "N-major": element (k, n) lives at srcT[n*ldT + k]
__device__ inline v16h load_b_fragT(const _Float16* __restrict__ srcT, int ldT,
                                    int n0, int k0) {
  const int lane = threadIdx.x & 31;
  const int g = lane >> 4;
  const _Float16* col = srcT + (n0 + (lane & 15)) * ldT + k0 + (g << 3);
  v16h b;
#pragma unroll
  for (int v = 0; v < 8; ++v) {
    h2 p = *(const h2*)(col + ((v >> 2) << 4) + ((v & 3) << 1));
    b[2 * v] = p.x;
    b[2 * v + 1] = p.y;
  }
  return b;
}

__device__ inline void store_d_f16(_Float16* __restrict__ dst, int ld, int m0,
                                   int n0, const v8f& c) {
  const int lane = threadIdx.x & 31;
  const int n = n0 + (lane & 15);
  const int mb = m0 + ((lane >> 4) << 3);
#pragma unroll
  for (int r = 0; r < 8; ++r) dst[(mb + r) * ld + n] = (_Float16)c[r];
}

// transposed store: dst[n][m]  (used to build V^T for the P*V GEMM)
__device__ inline void store_dT_f16(_Float16* __restrict__ dst, int ld, int m0,
                                    int n0, const v8f& c) {
  const int lane = threadIdx.x & 31;
  const int n = n0 + (lane & 15);
  const int mb = m0 + ((lane >> 4) << 3);
#pragma unroll
  for (int r = 0; r < 8; ++r) dst[n * ld + (mb + r)] = (_Float16)c[r];
}

// ---------------------------------------------------------------------------
// Kernel 0: repack one axis' attention weights f32 -> f16, transposed so the
// WMMA B-fragment (k,k+1) pair is a single contiguous 32-bit load.
//   wqT/wkT/wvT : [256][64]   (hidden-major, C contiguous)
//   woT         : [64][256]   (C-major, hidden contiguous)
// ---------------------------------------------------------------------------
__global__ void cvt_weights_kernel(const float* __restrict__ wq,
                                   const float* __restrict__ wkv,
                                   const float* __restrict__ wo,
                                   _Float16* __restrict__ wqT,
                                   _Float16* __restrict__ wkT,
                                   _Float16* __restrict__ wvT,
                                   _Float16* __restrict__ woT) {
  const int i = blockIdx.x * blockDim.x + threadIdx.x;
  if (i >= HID_ * C_) return;
  const int n = i >> 6, c = i & 63;               // [256][64]
  wqT[i] = (_Float16)wq[c * HID_ + n];
  wkT[i] = (_Float16)wkv[c * (2 * HID_) + n];
  wvT[i] = (_Float16)wkv[c * (2 * HID_) + HID_ + n];
  const int n2 = i >> 8, k2 = i & 255;            // [64][256]
  woT[i] = (_Float16)wo[k2 * C_ + n2];
}

// ---------------------------------------------------------------------------
// 3x3 SAME conv, Cin=Cout=64, 16x16 output tile per block, halo in LDS.
// MODE 0: in = x (NCHW f32) -> x1 (NCHW f32) and relu(x1) (BHWC f16)
// MODE 1: in = att (BHWC f32, relu applied on load) -> relu(x1 + conv) NCHW
// ---------------------------------------------------------------------------
template <int MODE>
__global__ __launch_bounds__(256) void conv3x3_kernel(
    const float* __restrict__ in, const float* __restrict__ w,
    const float* __restrict__ bias, const float* __restrict__ x1in,
    float* __restrict__ out_f32, _Float16* __restrict__ out_f16) {
  extern __shared__ float tile[];  // [18*18][64]
  const int blk = blockIdx.x;
  const int bx = blk & 7, by = (blk >> 3) & 7, b = blk >> 6;
  const int x0 = bx << 4, y0 = by << 4;
  const int tid = threadIdx.x;

  for (int idx = tid; idx < 18 * 18 * 64; idx += 256) {
    const int c = idx & 63;
    const int p = idx >> 6;
    const int xx = p % 18, yy = p / 18;
    const int gx = x0 + xx - 1, gy = y0 + yy - 1;
    float v = 0.0f;
    if (gx >= 0 && gx < HW_ && gy >= 0 && gy < HW_) {
      if (MODE == 0) {
        v = in[(((long)b * C_ + c) * HW_ + gy) * HW_ + gx];
      } else {
        v = in[(((long)b * HW_ + gy) * HW_ + gx) * C_ + c];
        v = fmaxf(v, 0.0f);  // relu before conv2
      }
    }
    tile[p * 64 + c] = v;
  }
  __syncthreads();

  const int px = tid & 15, py = tid >> 4;
  float acc[64];
#pragma unroll
  for (int co = 0; co < 64; ++co) acc[co] = 0.0f;

#pragma clang loop unroll(disable)
  for (int tap = 0; tap < 9; ++tap) {
    const int dy = tap / 3, dx = tap % 3;
    const float* trow = &tile[((py + dy) * 18 + (px + dx)) * 64];
#pragma clang loop unroll(disable)
    for (int ci = 0; ci < 64; ++ci) {
      const float v = trow[ci];
      const float* wp = w + (long)ci * 9 + tap;  // w[co][ci][tap], stride 576
#pragma unroll
      for (int co = 0; co < 64; ++co)
        acc[co] = fmaf(v, wp[(long)co * 576], acc[co]);
    }
  }

  const int gx = x0 + px, gy = y0 + py;
#pragma unroll
  for (int co = 0; co < 64; ++co) {
    const float y = acc[co] + bias[co];
    const long nchw = (((long)b * C_ + co) * HW_ + gy) * HW_ + gx;
    if (MODE == 0) {
      out_f32[nchw] = y;
      out_f16[(((long)b * HW_ + gy) * HW_ + gx) * C_ + co] =
          (_Float16)fmaxf(y, 0.0f);
    } else {
      out_f32[nchw] = fmaxf(x1in[nchw] + y, 0.0f);
    }
  }
}

// ---------------------------------------------------------------------------
// Axial multi-head attention over one length-128 sequence per workgroup.
// 8 waves, all GEMMs via v_wmma_f32_16x16x32_f16, f32 accumulation.
// LDS strides carry +8/+4 padding to dodge bank conflicts.
// ---------------------------------------------------------------------------
#define XS_LD  72
#define QK_LD  40
#define VT_LD  136
#define SF_LD  132
#define PS_LD  136
#define AC_LD  68

__global__ __launch_bounds__(256) void axial_attn_kernel(
    const _Float16* __restrict__ xf16,  // BHWC f16 (relu(conv1))
    const _Float16* __restrict__ wqT, const _Float16* __restrict__ wkT,
    const _Float16* __restrict__ wvT, const _Float16* __restrict__ woT,
    const float* __restrict__ wob, float* __restrict__ att,
    int strideFix, int tStride, int accumulate) {
  extern __shared__ char smem[];
  float* Sf = (float*)smem;                       // [128][132] f32 scores
  float* Acc = Sf + 128 * SF_LD;                  // [128][68]  f32 output acc
  _Float16* Xs = (_Float16*)(Acc + 128 * AC_LD);  // [128][72]
  _Float16* Qs = Xs + 128 * XS_LD;                // [128][40]
  _Float16* Ks = Qs + 128 * QK_LD;                // [128][40]
  _Float16* Os = Ks + 128 * QK_LD;                // [128][40]
  _Float16* Vt = Os + 128 * QK_LD;                // [32][136]  V^T
  _Float16* Ps = Vt + 32 * VT_LD;                 // [128][136] probs f16

  const int blk = blockIdx.x;
  const int b = blk >> 7;
  const int f = blk & 127;
  const long base = (long)b * (HW_ * HW_ * C_) + (long)f * strideFix;
  const int tid = threadIdx.x;
  const int wave = tid >> 5;

  // Pull the (hot, shared) f16 weight matrices toward the cache hierarchy.
  if (tid < 32) {
    __builtin_prefetch(wqT, 0, 1);
    __builtin_prefetch(wkT, 0, 1);
    __builtin_prefetch(wvT, 0, 1);
    __builtin_prefetch(woT, 0, 1);
  }

  // Stage the sequence [T=128][C=64] into LDS; seed accumulator with bias.
#if USE_ASYNC_LDS
  // 128 rows x 8 x 16B async copies (ASYNCcnt path, no VGPR data movement)
  for (int idx = tid; idx < 128 * 8; idx += 256) {
    const int t = idx >> 3, part = idx & 7;
    const _Float16* g = xf16 + base + (long)t * tStride + part * 8;
    _Float16* l = Xs + t * XS_LD + part * 8;
    __builtin_amdgcn_global_load_async_to_lds_b128(
        (__attribute__((address_space(1))) v4i*)(void*)g,
        (__attribute__((address_space(3))) v4i*)l, 0, 0);
  }
  for (int idx = tid; idx < 128 * 64; idx += 256) {
    const int t = idx >> 6, c = idx & 63;
    Acc[t * AC_LD + c] = wob[c];
  }
  __builtin_amdgcn_s_wait_asynccnt(0);
#else
  for (int idx = tid; idx < 128 * 64; idx += 256) {
    const int t = idx >> 6, c = idx & 63;
    Xs[t * XS_LD + c] = xf16[base + (long)t * tStride + c];
    Acc[t * AC_LD + c] = wob[c];
  }
#endif
  __syncthreads();

  for (int h0 = 0; h0 < 8; ++h0) {
    // ---- Q/K/V projections: [128,64] x [64,32] ; wave-uniform loops ----
    for (int t2 = wave; t2 < 16; t2 += 8) {  // Q
      const int mt = t2 >> 1, nt = t2 & 1;
      v8f c = {};
#pragma unroll
      for (int ks = 0; ks < 2; ++ks) {
        v16h a = load_a_frag(Xs, XS_LD, mt * 16, ks * 32);
        v16h bm = load_b_fragT(wqT, 64, h0 * 32 + nt * 16, ks * 32);
        c = __builtin_amdgcn_wmma_f32_16x16x32_f16(false, a, false, bm,
                                                   (short)0, c, false, false);
      }
      store_d_f16(Qs, QK_LD, mt * 16, nt * 16, c);
    }
    for (int t2 = wave; t2 < 16; t2 += 8) {  // K
      const int mt = t2 >> 1, nt = t2 & 1;
      v8f c = {};
#pragma unroll
      for (int ks = 0; ks < 2; ++ks) {
        v16h a = load_a_frag(Xs, XS_LD, mt * 16, ks * 32);
        v16h bm = load_b_fragT(wkT, 64, h0 * 32 + nt * 16, ks * 32);
        c = __builtin_amdgcn_wmma_f32_16x16x32_f16(false, a, false, bm,
                                                   (short)0, c, false, false);
      }
      store_d_f16(Ks, QK_LD, mt * 16, nt * 16, c);
    }
    for (int t2 = wave; t2 < 16; t2 += 8) {  // V (stored transposed)
      const int mt = t2 >> 1, nt = t2 & 1;
      v8f c = {};
#pragma unroll
      for (int ks = 0; ks < 2; ++ks) {
        v16h a = load_a_frag(Xs, XS_LD, mt * 16, ks * 32);
        v16h bm = load_b_fragT(wvT, 64, h0 * 32 + nt * 16, ks * 32);
        c = __builtin_amdgcn_wmma_f32_16x16x32_f16(false, a, false, bm,
                                                   (short)0, c, false, false);
      }
      store_dT_f16(Vt, VT_LD, mt * 16, nt * 16, c);
    }
    __syncthreads();

    // ---- scores S = Q * K^T : [128,32]x[32,128], 64 tiles, 1 k-step ----
    for (int job = wave; job < 64; job += 8) {
      const int mt = job >> 3, jt = job & 7;
      v16h a = load_a_frag(Qs, QK_LD, mt * 16, 0);
      v16h bm = load_b_fragT(Ks, QK_LD, jt * 16, 0);  // B[e][j] = K[j][e]
      v8f c = {};
      c = __builtin_amdgcn_wmma_f32_16x16x32_f16(false, a, false, bm, (short)0,
                                                 c, false, false);
      const int lane = tid & 31;
      const int n = jt * 16 + (lane & 15);
      const int mb = mt * 16 + ((lane >> 4) << 3);
#pragma unroll
      for (int r = 0; r < 8; ++r) Sf[(mb + r) * SF_LD + n] = c[r];
    }
    __syncthreads();

    // ---- row softmax (scale = 32^-0.5); exp recomputed, no scratch ----
    if (tid < 128) {
      const float scale = 0.17677669529663689f;
      const float* row = Sf + tid * SF_LD;
      float mx = -3.0e38f;
#pragma clang loop unroll(disable)
      for (int j = 0; j < 128; ++j) mx = fmaxf(mx, row[j]);
      float s = 0.0f;
#pragma clang loop unroll(disable)
      for (int j = 0; j < 128; ++j) s += __expf((row[j] - mx) * scale);
      const float inv = 1.0f / s;
#pragma clang loop unroll(disable)
      for (int j = 0; j < 128; ++j)
        Ps[tid * PS_LD + j] = (_Float16)(__expf((row[j] - mx) * scale) * inv);
    }
    __syncthreads();

    // ---- O = P * V : [128,128]x[128,32], 16 tiles, 4 k-steps ----
    for (int job = wave; job < 16; job += 8) {
      const int mt = job >> 1, nt = job & 1;
      v8f c = {};
#pragma unroll
      for (int ks = 0; ks < 4; ++ks) {
        v16h a = load_a_frag(Ps, PS_LD, mt * 16, ks * 32);
        v16h bm = load_b_fragT(Vt, VT_LD, nt * 16, ks * 32);  // B[j][e]=V[j][e]
        c = __builtin_amdgcn_wmma_f32_16x16x32_f16(false, a, false, bm,
                                                   (short)0, c, false, false);
      }
      store_d_f16(Os, QK_LD, mt * 16, nt * 16, c);
    }
    __syncthreads();

    // ---- per-head out-proj: Acc += O_h * Wo[h*32:(h+1)*32, :] ----
    for (int job = wave; job < 32; job += 8) {
      const int mt = job >> 2, nt = job & 3;
      v16h a = load_a_frag(Os, QK_LD, mt * 16, 0);
      v16h bm = load_b_fragT(woT, HID_, nt * 16, h0 * 32);
      v8f c = {};
      c = __builtin_amdgcn_wmma_f32_16x16x32_f16(false, a, false, bm, (short)0,
                                                 c, false, false);
      const int lane = tid & 31;
      const int n = nt * 16 + (lane & 15);
      const int mb = mt * 16 + ((lane >> 4) << 3);
#pragma unroll
      for (int r = 0; r < 8; ++r) Acc[(mb + r) * AC_LD + n] += c[r];
    }
    __syncthreads();
  }

  // write (or accumulate) the [128][64] result back to the BHWC att buffer
  for (int idx = tid; idx < 128 * 64; idx += 256) {
    const int t = idx >> 6, c = idx & 63;
    const long addr = base + (long)t * tStride + c;
    const float v = Acc[t * AC_LD + c];
    if (accumulate) att[addr] += v;
    else            att[addr] = v;
  }
}

// ---------------------------------------------------------------------------
// host-side orchestration
// ---------------------------------------------------------------------------
extern "C" void kernel_launch(void* const* d_in, const int* in_sizes, int n_in,
                              void* d_out, int out_size, void* d_ws,
                              size_t ws_size, hipStream_t stream) {
  const float* x      = (const float*)d_in[0];
  const float* conv_w = (const float*)d_in[1];
  const float* conv_b = (const float*)d_in[2];
  const float* wq_h   = (const float*)d_in[3];
  const float* wkv_h  = (const float*)d_in[4];
  const float* wo_h   = (const float*)d_in[5];
  const float* wob_h  = (const float*)d_in[6];
  const float* wq_w   = (const float*)d_in[7];
  const float* wkv_w  = (const float*)d_in[8];
  const float* wo_w   = (const float*)d_in[9];
  const float* wob_w  = (const float*)d_in[10];
  float* out = (float*)d_out;

  char* ws = (char*)d_ws;
  size_t off = 0;
  auto carve = [&](size_t bytes) -> void* {
    void* p = ws + off;
    off = (off + bytes + 255) & ~(size_t)255;
    return p;
  };
  float*     x1   = (float*)carve((size_t)NELEM * 4);       // conv1 out, NCHW
  float*     att  = (float*)carve((size_t)NELEM * 4);       // a_h + a_w, BHWC
  _Float16*  x2h  = (_Float16*)carve((size_t)NELEM * 2);    // relu(x1), BHWC
  _Float16*  wqTh = (_Float16*)carve(HID_ * C_ * 2);
  _Float16*  wkTh = (_Float16*)carve(HID_ * C_ * 2);
  _Float16*  wvTh = (_Float16*)carve(HID_ * C_ * 2);
  _Float16*  woTh = (_Float16*)carve(HID_ * C_ * 2);
  _Float16*  wqTw = (_Float16*)carve(HID_ * C_ * 2);
  _Float16*  wkTw = (_Float16*)carve(HID_ * C_ * 2);
  _Float16*  wvTw = (_Float16*)carve(HID_ * C_ * 2);
  _Float16*  woTw = (_Float16*)carve(HID_ * C_ * 2);

  // 0) repack weights to f16 transposed layouts
  cvt_weights_kernel<<<64, 256, 0, stream>>>(wq_h, wkv_h, wo_h, wqTh, wkTh,
                                             wvTh, woTh);
  cvt_weights_kernel<<<64, 256, 0, stream>>>(wq_w, wkv_w, wo_w, wqTw, wkTw,
                                             wvTw, woTw);

  // 1) conv1: x1 = conv(x)+b ; x2h = relu(x1) as f16 BHWC
  const size_t convSmem = 18 * 18 * 64 * sizeof(float);  // 82944 B
  conv3x3_kernel<0><<<B_ * 8 * 8, 256, convSmem, stream>>>(x, conv_w, conv_b,
                                                           nullptr, x1, x2h);

  // 2) axial attention. LDS: scores f32 + acc f32 + X/Q/K/O/V^T/P f16
  const size_t attnSmem =
      (size_t)(128 * SF_LD + 128 * AC_LD) * 4 +
      (size_t)(128 * XS_LD + 3 * 128 * QK_LD + 32 * VT_LD + 128 * PS_LD) * 2;
  // along W: blk = b*128 + h ; seq stride 64 (contiguous rows)
  axial_attn_kernel<<<B_ * HW_, 256, attnSmem, stream>>>(
      x2h, wqTw, wkTw, wvTw, woTw, wob_w, att,
      /*strideFix=*/HW_ * C_, /*tStride=*/C_, /*accumulate=*/0);
  // along H: blk = b*128 + w ; seq stride W*C
  axial_attn_kernel<<<B_ * HW_, 256, attnSmem, stream>>>(
      x2h, wqTh, wkTh, wvTh, woTh, wob_h, att,
      /*strideFix=*/C_, /*tStride=*/HW_ * C_, /*accumulate=*/1);

  // 3) conv2 over relu(att), residual with x1, final relu -> d_out (NCHW)
  conv3x3_kernel<1><<<B_ * 8 * 8, 256, convSmem, stream>>>(att, conv_w, conv_b,
                                                           x1, out, nullptr);
}
